// NEMHSA_22806276342191
// MI455X (gfx1250) — matrix operand order repaired and checked by reading, not compiled, not run on hardware
//
#include <hip/hip_runtime.h>
#include <hip/hip_bf16.h>

typedef __attribute__((ext_vector_type(16))) _Float16 v16h;
typedef __attribute__((ext_vector_type(8)))  _Float16 v8h;
typedef __attribute__((ext_vector_type(8)))  float    v8f;

#define NB  4
#define NT  2048
#define ND  1024
#define NE  4
#define NH  8
#define NHD 128
#define KPE 512

union AF { v16h v; v8h h[2]; };

static __device__ __forceinline__ v8f wmma16(v16h a, v16h b, v8f c) {
  return __builtin_amdgcn_wmma_f32_16x16x32_f16(false, a, false, b, (short)0, c,
                                                false, false);
}

// Async memory->LDS copy of 16B, per-lane (CDNA5, tracked by ASYNCcnt).
static __device__ __forceinline__ void async_ld128(void* lds, const void* gaddr) {
  unsigned l = (unsigned)(unsigned long long)lds;       // low 32 bits = LDS addr
  unsigned long long g = (unsigned long long)gaddr;
  asm volatile("global_load_async_to_lds_b128 %0, %1, off"
               :: "v"(l), "v"(g) : "memory");
}
static __device__ __forceinline__ void async_wait0() {
  asm volatile("s_wait_asynccnt 0x0" ::: "memory");
}

// ---------------------------------------------------------------- f32 -> f16
__global__ void nem_cvt_f16(const float* __restrict__ s, _Float16* __restrict__ d, int n) {
  int i = blockIdx.x * blockDim.x + threadIdx.x;
  if (i < n) d[i] = (_Float16)s[i];
}

// ---------------------------------------------------- sequential top-k route
__global__ void nem_route(const float* __restrict__ prob_in, int* __restrict__ perm) {
  __shared__ float pbuf[NT * NE];
  __shared__ float key[NT];
  __shared__ int   val[NT];
  const int b = blockIdx.x;
  const int tid = threadIdx.x;
  for (int i = tid; i < NT * NE; i += 1024) pbuf[i] = prob_in[(size_t)b * NT * NE + i];
  __syncthreads();
  for (int ei = 0; ei < NE; ++ei) {
    for (int i = tid; i < NT; i += 1024) { key[i] = pbuf[i * NE + ei]; val[i] = i; }
    __syncthreads();
    for (int ks = 2; ks <= NT; ks <<= 1) {
      for (int j = ks >> 1; j > 0; j >>= 1) {
        for (int base = 0; base < NT; base += 1024) {
          int i = base + tid;
          int ixj = i ^ j;
          if (ixj > i) {
            bool descSeg = ((i & ks) == 0);
            float ki = key[i], kj = key[ixj];
            bool doswap = descSeg ? (ki < kj) : (ki > kj);
            if (doswap) {
              key[i] = kj; key[ixj] = ki;
              int t = val[i]; val[i] = val[ixj]; val[ixj] = t;
            }
          }
        }
        __syncthreads();
      }
    }
    if (tid < KPE) perm[(size_t)b * NT + ei * KPE + tid] = val[tid];
    __syncthreads();
    if (tid < KPE) {
      int tok = val[tid];
      for (int c = 0; c < NE; ++c) pbuf[tok * NE + c] = 0.0f;
    }
    __syncthreads();
  }
}

// -------------------------------------------------- gather probs by a perm
__global__ void nem_gather_probs(const float* __restrict__ src, const int* __restrict__ perm,
                                 float* __restrict__ dst) {
  int idx = blockIdx.x * blockDim.x + threadIdx.x;
  if (idx >= NB * NT * NE) return;
  int e = idx & (NE - 1);
  int i = (idx >> 2) & (NT - 1);
  int b = idx >> 13;
  dst[idx] = src[((size_t)b * NT + perm[b * NT + i]) * NE + e];
}

// ----------------------------------------------- gathered LayerNorm -> f16
__global__ void nem_ln(const float* __restrict__ x, const int* __restrict__ perm,
                       const float* __restrict__ g, const float* __restrict__ bb,
                       _Float16* __restrict__ out) {
  __shared__ float red[256];
  const int p = blockIdx.x, b = blockIdx.y;
  const int tid = threadIdx.x;
  const int tok = perm[b * NT + p];
  const float* row = x + ((size_t)b * NT + tok) * ND;
  float v[4]; float s = 0.0f;
  for (int i = 0; i < 4; ++i) { v[i] = row[tid * 4 + i]; s += v[i]; }
  red[tid] = s; __syncthreads();
  for (int o = 128; o > 0; o >>= 1) { if (tid < o) red[tid] += red[tid + o]; __syncthreads(); }
  float mu = red[0] * (1.0f / ND);
  __syncthreads();
  float s2 = 0.0f;
  for (int i = 0; i < 4; ++i) { float d = v[i] - mu; s2 += d * d; }
  red[tid] = s2; __syncthreads();
  for (int o = 128; o > 0; o >>= 1) { if (tid < o) red[tid] += red[tid + o]; __syncthreads(); }
  float inv = rsqrtf(red[0] * (1.0f / ND) + 1e-5f);
  _Float16* orow = out + ((size_t)b * NT + p) * ND;
  for (int i = 0; i < 4; ++i) {
    int c = tid * 4 + i;
    orow[c] = (_Float16)((v[i] - mu) * inv * g[c] + bb[c]);
  }
}

// ------------------------------------------ fused Q/K/V WMMA GEMM (f16 in/out)
__global__ void nem_qkv(const _Float16* __restrict__ xln,
                        const _Float16* __restrict__ qw, const _Float16* __restrict__ kw,
                        const _Float16* __restrict__ vw,
                        const float* __restrict__ qb, const float* __restrict__ kb,
                        const float* __restrict__ vb,
                        _Float16* __restrict__ Q, _Float16* __restrict__ K,
                        _Float16* __restrict__ V) {
  __shared__ __align__(16) _Float16 aT[64][32];
  __shared__ __align__(16) _Float16 bq[64][32];
  __shared__ __align__(16) _Float16 bk[64][32];
  __shared__ __align__(16) _Float16 bv[64][32];
  const int b = blockIdx.z, m0 = blockIdx.x * 64, n0 = blockIdx.y * 64;
  const int m = ND >> (m0 / KPE);  // nested width for this expert segment
  const int tid = threadIdx.x, lane = tid & 31, wave = tid >> 5;
  const int li = lane & 15, hi = lane >> 4;
  v8f zero8; for (int r = 0; r < 8; ++r) zero8[r] = 0.0f;
  v8f accq[4], acck[4], accv[4];
  for (int t = 0; t < 4; ++t) { accq[t] = zero8; acck[t] = zero8; accv[t] = zero8; }

  const int r = tid >> 1, hb = (tid & 1) * 16;
  for (int k0 = 0; k0 < m; k0 += 32) {
    __syncthreads();
    {  // async memory->LDS tile stage (no VGPR round trip)
      const _Float16* as  = xln + ((size_t)b * NT + m0 + r) * ND + k0 + hb;
      const _Float16* qs  = qw + (size_t)(n0 + r) * ND + k0 + hb;
      const _Float16* ks2 = kw + (size_t)(n0 + r) * ND + k0 + hb;
      const _Float16* vs  = vw + (size_t)(n0 + r) * ND + k0 + hb;
      async_ld128(&aT[r][hb], as);      async_ld128(&aT[r][hb + 8], as + 8);
      async_ld128(&bq[r][hb], qs);      async_ld128(&bq[r][hb + 8], qs + 8);
      async_ld128(&bk[r][hb], ks2);     async_ld128(&bk[r][hb + 8], ks2 + 8);
      async_ld128(&bv[r][hb], vs);      async_ld128(&bv[r][hb + 8], vs + 8);
    }
    async_wait0();
    __syncthreads();
    AF ua;
    const _Float16* ar = &aT[wave * 16 + li][0];
    ua.h[0] = *(const v8h*)(ar + hi * 16);
    ua.h[1] = *(const v8h*)(ar + hi * 16 + 8);
    for (int nt = 0; nt < 4; ++nt) {
      int brow = nt * 16 + li;
      AF uq, uk, uv;
      uq.h[0] = *(const v8h*)&bq[brow][hi * 16]; uq.h[1] = *(const v8h*)&bq[brow][hi * 16 + 8];
      uk.h[0] = *(const v8h*)&bk[brow][hi * 16]; uk.h[1] = *(const v8h*)&bk[brow][hi * 16 + 8];
      uv.h[0] = *(const v8h*)&bv[brow][hi * 16]; uv.h[1] = *(const v8h*)&bv[brow][hi * 16 + 8];
      accq[nt] = wmma16(ua.v, uq.v, accq[nt]);
      acck[nt] = wmma16(ua.v, uk.v, acck[nt]);
      accv[nt] = wmma16(ua.v, uv.v, accv[nt]);
    }
  }
  for (int nt = 0; nt < 4; ++nt) {
    int col = n0 + nt * 16 + li;
    float qbv = qb[col], kbv = kb[col], vbv = vb[col];
    for (int rr = 0; rr < 8; ++rr) {
      int row = m0 + wave * 16 + 8 * hi + rr;
      size_t o = ((size_t)b * NT + row) * ND + col;
      Q[o] = (_Float16)(accq[nt][rr] + qbv);
      K[o] = (_Float16)(acck[nt][rr] + kbv);
      V[o] = (_Float16)(accv[nt][rr] + vbv);
    }
  }
}

// --------------------------------------- flash attention, WMMA, online softmax
__global__ void nem_attn(const _Float16* __restrict__ Q, const _Float16* __restrict__ K,
                         const _Float16* __restrict__ V, _Float16* __restrict__ Ao) {
  __shared__ __align__(16) _Float16 kT[32][128];   // [j][d]
  __shared__ __align__(16) _Float16 vT[128][32];   // [d][j] (transposed)
  __shared__ float    sS[8][16][32];               // per-wave raw scores
  __shared__ __align__(16) _Float16 pT[8][16][32]; // per-wave P (f16)
  __shared__ float mst[8][16], lst[8][16], alph[8][16];
  const int b = blockIdx.z, h = blockIdx.y, q0 = blockIdx.x * 128;
  const int tid = threadIdx.x, lane = tid & 31, wave = tid >> 5;
  const int li = lane & 15, hi = lane >> 4;
  const float scale = 0.03125f;  // D^-0.5 = 1/32

  if (lane < 16) { mst[wave][lane] = -3.0e30f; lst[wave][lane] = 0.0f; }

  AF aq[4];
  const _Float16* qrow = Q + ((size_t)b * NT + q0 + wave * 16 + li) * ND + h * NHD;
  for (int c = 0; c < 4; ++c) {
    aq[c].h[0] = *(const v8h*)(qrow + c * 32 + hi * 16);
    aq[c].h[1] = *(const v8h*)(qrow + c * 32 + hi * 16 + 8);
  }
  v8f zero8; for (int r = 0; r < 8; ++r) zero8[r] = 0.0f;
  v8f co[8]; for (int t = 0; t < 8; ++t) co[t] = zero8;

  for (int kv0 = 0; kv0 < NT; kv0 += 32) {
    __syncthreads();
    {  // K tile via async memory->LDS copy
      int kr = tid >> 3, cb = (tid & 7) * 16;
      const _Float16* src = K + ((size_t)b * NT + kv0 + kr) * ND + h * NHD + cb;
      async_ld128(&kT[kr][cb], src);
      async_ld128(&kT[kr][cb + 8], src + 8);
    }
    {  // V tile transposed into [d][j] (scatter: stays on VGPR path)
      int j = tid >> 3, db = (tid & 7) * 16;
      const _Float16* src = V + ((size_t)b * NT + kv0 + j) * ND + h * NHD + db;
      for (int t = 0; t < 16; ++t) vT[db + t][j] = src[t];
    }
    async_wait0();
    __syncthreads();

    // S = Q @ K^T (16 x 32) via 2 C-frags x 4 K-chunks
    v8f s0 = zero8, s1 = zero8;
    for (int c = 0; c < 4; ++c) {
      AF b0, b1;
      const _Float16* kr0 = &kT[li][c * 32 + hi * 16];
      b0.h[0] = *(const v8h*)kr0; b0.h[1] = *(const v8h*)(kr0 + 8);
      const _Float16* kr1 = &kT[16 + li][c * 32 + hi * 16];
      b1.h[0] = *(const v8h*)kr1; b1.h[1] = *(const v8h*)(kr1 + 8);
      s0 = wmma16(aq[c].v, b0.v, s0);
      s1 = wmma16(aq[c].v, b1.v, s1);
    }
    for (int r = 0; r < 8; ++r) {
      int row = r + 8 * hi;
      sS[wave][row][li] = s0[r] * scale;
      sS[wave][row][16 + li] = s1[r] * scale;
    }
    __syncthreads();
    if (lane < 16) {  // per-row online softmax update
      int row = lane;
      float mo = mst[wave][row], mx = mo;
      for (int j = 0; j < 32; ++j) mx = fmaxf(mx, sS[wave][row][j]);
      float a = __expf(mo - mx), ls = 0.0f;
      for (int j = 0; j < 32; ++j) {
        float p = __expf(sS[wave][row][j] - mx);
        pT[wave][row][j] = (_Float16)p;
        ls += p;
      }
      mst[wave][row] = mx;
      lst[wave][row] = lst[wave][row] * a + ls;
      alph[wave][row] = a;
    }
    __syncthreads();

    // O^T += V^T @ P^T  (per-lane q-row => scalar rescale)
    float a = alph[wave][li];
    AF bp;
    const _Float16* pr = &pT[wave][li][hi * 16];
    bp.h[0] = *(const v8h*)pr; bp.h[1] = *(const v8h*)(pr + 8);
    for (int dt = 0; dt < 8; ++dt) {
      for (int r = 0; r < 8; ++r) co[dt][r] *= a;
      AF av;
      const _Float16* vr = &vT[dt * 16 + li][0];
      av.h[0] = *(const v8h*)(vr + hi * 16);
      av.h[1] = *(const v8h*)(vr + hi * 16 + 8);
      co[dt] = wmma16(av.v, bp.v, co[dt]);
    }
  }
  __syncthreads();
  float linv = 1.0f / lst[wave][li];
  _Float16* orow = Ao + ((size_t)b * NT + q0 + wave * 16 + li) * ND + h * NHD;
  for (int dt = 0; dt < 8; ++dt) {
    _Float16 tmp[8];
    for (int r = 0; r < 8; ++r) tmp[r] = (_Float16)(co[dt][r] * linv);
    *(v8h*)(orow + dt * 16 + 8 * hi) = *(const v8h*)tmp;
  }
}

// ------------------------- O-projection (m x m) + residual + pad, f32 output
__global__ void nem_oproj(const _Float16* __restrict__ Ah, const _Float16* __restrict__ ow,
                          const float* __restrict__ ob, const float* __restrict__ x,
                          const int* __restrict__ perm1, const int* __restrict__ perm2,
                          float* __restrict__ out) {
  __shared__ int rows2[64], toks[64];
  __shared__ __align__(16) _Float16 aT[64][32];
  __shared__ __align__(16) _Float16 bT[64][32];
  const int b = blockIdx.z, m0 = blockIdx.x * 64, n0 = blockIdx.y * 64;
  const int m = ND >> (m0 / KPE);
  const int tid = threadIdx.x, lane = tid & 31, wave = tid >> 5;
  const int li = lane & 15, hi = lane >> 4;
  if (tid < 64) {
    int i2 = perm2[b * NT + m0 + tid];
    rows2[tid] = i2;
    toks[tid] = perm1[b * NT + i2];
  }
  __syncthreads();
  if (n0 >= m) {  // pad region: residual only (uniform per block)
    for (int idx = tid; idx < 64 * 64; idx += 128) {
      int rr = idx >> 6, c = idx & 63;
      out[((size_t)b * NT + m0 + rr) * ND + n0 + c] =
          x[((size_t)b * NT + toks[rr]) * ND + n0 + c];
    }
    return;
  }
  v8f zero8; for (int r = 0; r < 8; ++r) zero8[r] = 0.0f;
  v8f acc[4]; for (int t = 0; t < 4; ++t) acc[t] = zero8;
  const int r = tid >> 1, hb = (tid & 1) * 16;
  for (int k0 = 0; k0 < m; k0 += 32) {
    __syncthreads();
    {
      const _Float16* as = Ah + ((size_t)b * NT + rows2[r]) * ND + k0 + hb;
      const _Float16* bs = ow + (size_t)(n0 + r) * ND + k0 + hb;
      async_ld128(&aT[r][hb], as);  async_ld128(&aT[r][hb + 8], as + 8);
      async_ld128(&bT[r][hb], bs);  async_ld128(&bT[r][hb + 8], bs + 8);
    }
    async_wait0();
    __syncthreads();
    AF ua;
    const _Float16* ar = &aT[wave * 16 + li][0];
    ua.h[0] = *(const v8h*)(ar + hi * 16);
    ua.h[1] = *(const v8h*)(ar + hi * 16 + 8);
    for (int nt = 0; nt < 4; ++nt) {
      AF ub;
      const _Float16* br = &bT[nt * 16 + li][hi * 16];
      ub.h[0] = *(const v8h*)br; ub.h[1] = *(const v8h*)(br + 8);
      acc[nt] = wmma16(ua.v, ub.v, acc[nt]);
    }
  }
  for (int nt = 0; nt < 4; ++nt) {
    int col = n0 + nt * 16 + li;
    float obv = ob[col];
    for (int rr = 0; rr < 8; ++rr) {
      int lr = wave * 16 + 8 * hi + rr;
      out[((size_t)b * NT + m0 + lr) * ND + col] =
          acc[nt][rr] + obv + x[((size_t)b * NT + toks[lr]) * ND + col];
    }
  }
}

// ============================================================================
extern "C" void kernel_launch(void* const* d_in, const int* in_sizes, int n_in,
                              void* d_out, int out_size, void* d_ws, size_t ws_size,
                              hipStream_t stream) {
  (void)in_sizes; (void)n_in; (void)out_size; (void)ws_size;
  const float* x    = (const float*)d_in[0];
  const float* rp   = (const float*)d_in[1];
  const float* ln_g = (const float*)d_in[2];
  const float* ln_b = (const float*)d_in[3];
  const float* q_w  = (const float*)d_in[4];
  const float* q_b  = (const float*)d_in[5];
  const float* k_w  = (const float*)d_in[6];
  const float* k_b  = (const float*)d_in[7];
  const float* v_w  = (const float*)d_in[8];
  const float* v_b  = (const float*)d_in[9];
  const float* o_w  = (const float*)d_in[10];
  const float* o_b  = (const float*)d_in[11];
  float* out = (float*)d_out;

  char* ws = (char*)d_ws;
  size_t off = 0;
  auto carve = [&](size_t bytes) { void* p = ws + off; off += (bytes + 255) & ~(size_t)255; return p; };
  int*       perm1 = (int*)carve((size_t)NB * NT * 4);
  int*       perm2 = (int*)carve((size_t)NB * NT * 4);
  float*     newp  = (float*)carve((size_t)NB * NT * NE * 4);
  _Float16*  xln   = (_Float16*)carve((size_t)NB * NT * ND * 2);
  _Float16*  Qh    = (_Float16*)carve((size_t)NB * NT * ND * 2);
  _Float16*  Kh    = (_Float16*)carve((size_t)NB * NT * ND * 2);
  _Float16*  Vh    = (_Float16*)carve((size_t)NB * NT * ND * 2);
  _Float16*  Ah    = (_Float16*)carve((size_t)NB * NT * ND * 2);
  _Float16*  qwh   = (_Float16*)carve((size_t)ND * ND * 2);
  _Float16*  kwh   = (_Float16*)carve((size_t)ND * ND * 2);
  _Float16*  vwh   = (_Float16*)carve((size_t)ND * ND * 2);
  _Float16*  owh   = (_Float16*)carve((size_t)ND * ND * 2);

  const int nw = ND * ND;
  nem_cvt_f16<<<(nw + 255) / 256, 256, 0, stream>>>(q_w, qwh, nw);
  nem_cvt_f16<<<(nw + 255) / 256, 256, 0, stream>>>(k_w, kwh, nw);
  nem_cvt_f16<<<(nw + 255) / 256, 256, 0, stream>>>(v_w, vwh, nw);
  nem_cvt_f16<<<(nw + 255) / 256, 256, 0, stream>>>(o_w, owh, nw);

  nem_route<<<NB, 1024, 0, stream>>>(rp, perm1);
  nem_gather_probs<<<(NB * NT * NE + 255) / 256, 256, 0, stream>>>(rp, perm1, newp);
  nem_route<<<NB, 1024, 0, stream>>>(newp, perm2);

  nem_ln<<<dim3(NT, NB), 256, 0, stream>>>(x, perm1, ln_g, ln_b, xln);

  nem_qkv<<<dim3(NT / 64, ND / 64, NB), 128, 0, stream>>>(
      xln, qwh, kwh, vwh, q_b, k_b, v_b, Qh, Kh, Vh);

  nem_attn<<<dim3(NT / 128, NH, NB), 256, 0, stream>>>(Qh, Kh, Vh, Ah);

  nem_oproj<<<dim3(NT / 64, ND / 64, NB), 128, 0, stream>>>(
      Ah, owh, o_b, x, perm1, perm2, out);

  nem_gather_probs<<<(NB * NT * NE + 255) / 256, 256, 0, stream>>>(
      newp, perm2, out + (size_t)NB * NT * ND);
}
